// HunterModel_12927851561511
// MI455X (gfx1250) — matrix operand order repaired and checked by reading, not compiled, not run on hardware
//
#include <hip/hip_runtime.h>
#include <hip/hip_bf16.h>
#include <stdint.h>

// ---------------------------------------------------------------------------
// GCN forward for MI455X (gfx1250, wave32, WMMA).
//   h = relu(Ahat (x W1) + b1); h = relu(Ahat (h W2) + b2); h = relu(Ahat (h W3) + b3)
//   node = relu(h[idx] Wp + bp); out = concat(node Wa+ba, Wm+bm, Wg+bg, Wt+bt)
//
// Design notes (compile-only, reasoned from MI455X specs):
//  - Feature buffers are 100000x128 f32 = 51.2 MB -> fit in the 192 MB L2, so
//    the edge gather + scatter-add phase is L2-resident, not HBM bound.
//  - GEMMs (8.2 GFLOP total) run on v_wmma_f32_16x16x32_f16 (f16 in, f32 acc).
//  - GEMM prologue fuses relu + f32->f16 convert while staging A into LDS;
//    GEMM epilogue fuses the self-loop seed (z*dinv^2 + bias) so the only
//    remaining per-layer passes are the tiny weight pack and the edge scatter.
//  - Scatter is 4-wide per thread: one edge-index/norm fetch amortized over a
//    b128 z read and 4 global_atomic_add_f32.
// ---------------------------------------------------------------------------

#define GN 100000
#define GE 1600000
#define GDIN 64
#define GDH 128

typedef __attribute__((ext_vector_type(16))) _Float16 v16h;
typedef __attribute__((ext_vector_type(8)))  float    v8f;

// ---------------- degree / norm ----------------

__global__ void k_count(const int* __restrict__ ei, uint32_t* __restrict__ cnt) {
    int e = blockIdx.x * blockDim.x + threadIdx.x;
    if (e < GE) {
        __builtin_prefetch(&ei[GE + e + 4096], 0, 1);   // global_prefetch_b8
        int dst = ei[GE + e];
        atomicAdd(&cnt[dst], 1u);
    }
}

__global__ void k_dinv(const uint32_t* __restrict__ cnt, float* __restrict__ dinv) {
    int i = blockIdx.x * blockDim.x + threadIdx.x;
    if (i < GN) dinv[i] = rsqrtf((float)cnt[i] + 1.0f);  // +1 self-loop
}

// Pack W [K x 128] f32 row-major -> wh [128 x K] f16 (transposed), so the
// WMMA B fragment (per-column contiguous K) loads as contiguous b128s.
__global__ void k_pack_w(const float* __restrict__ W, _Float16* __restrict__ wh, int K) {
    int idx = blockIdx.x * blockDim.x + threadIdx.x;     // 128*K
    if (idx < GDH * K) {
        int n = idx / K, k = idx % K;
        wh[idx] = (_Float16)W[k * GDH + n];
    }
}

// ---------------- WMMA GEMM with fused relu-prologue + seed-epilogue --------
// z[N x 128]   = relu?(A)[N x K] * W[K x 128]          (f16 WMMA, f32 accum)
// agg[N x 128] = z * dinv^2 + bias                     (self-loop + bias seed)
// Block: 256 threads = 8 waves; block handles 16 rows, wave w -> cols [16w,16w+16).

__global__ void __launch_bounds__(256)
k_gemm(const float* __restrict__ A, int K, int applyRelu,
       const _Float16* __restrict__ wh, const float* __restrict__ dinv,
       const float* __restrict__ bias, float* __restrict__ z,
       float* __restrict__ agg, int n) {
    __shared__ uint64_t sA64[16 * GDH / 4];              // 16 rows x K halves, <=4 KB
    const _Float16* sA = (const _Float16*)sA64;

    const int row0 = blockIdx.x * 16;
    const int tid  = threadIdx.x;
    const int wv   = tid >> 5;
    const int lane = tid & 31;
    const int m    = lane & 15;       // A row within tile / B column within tile
    const int h    = lane >> 4;       // half-wave select

    // Stage A tile: read f32 (b128), fused relu, convert to f16, ds-store b64.
    const int pk_per_row = K / 4;                         // 4-half packets per row
    for (int idx = tid; idx < 16 * pk_per_row; idx += 256) {
        int r = idx / pk_per_row, q = idx % pk_per_row;
        int grow = row0 + r;
        float4 v = make_float4(0.f, 0.f, 0.f, 0.f);
        if (grow < n) v = ((const float4*)(A + (size_t)grow * K))[q];
        if (applyRelu) {
            v.x = fmaxf(v.x, 0.f); v.y = fmaxf(v.y, 0.f);
            v.z = fmaxf(v.z, 0.f); v.w = fmaxf(v.w, 0.f);
        }
        union { _Float16 hh[4]; uint64_t u; } pk;
        pk.hh[0] = (_Float16)v.x; pk.hh[1] = (_Float16)v.y;
        pk.hh[2] = (_Float16)v.z; pk.hh[3] = (_Float16)v.w;
        sA64[r * pk_per_row + q] = pk.u;
    }
    __syncthreads();

    const int col = (wv << 4) + m;                        // output column 0..127
    v8f c = {};

    for (int kb = 0; kb < K; kb += 32) {
        // A fragment: lanes 0-15 rows M=0..15; VGPR i -> K = kb + h*16 + i*2
        v16h a;
#pragma unroll
        for (int i = 0; i < 8; ++i) {
            int kk = kb + (h << 4) + (i << 1);
            a[2 * i]     = sA[m * K + kk];
            a[2 * i + 1] = sA[m * K + kk + 1];
        }
        // B fragment: lane = column, K range h*16..h*16+15; contiguous in wh^T.
        const _Float16* bp = wh + (size_t)col * K + kb + (h << 4);
        v16h b;
#pragma unroll
        for (int i = 0; i < 16; ++i) b[i] = bp[i];

        c = __builtin_amdgcn_wmma_f32_16x16x32_f16(
                /*neg_a=*/false, a, /*neg_b=*/false, b,
                /*c_mod=*/(short)0, c, /*reuse_a=*/false, /*reuse_b=*/false);
    }

    // D layout: lanes 0-15 VGPR r -> M=r; lanes 16-31 VGPR r -> M=8+r.
    const float bcol = bias[col];
#pragma unroll
    for (int r = 0; r < 8; ++r) {
        int grow = row0 + r + (h << 3);
        if (grow < n) {
            float val = c[r];
            float di  = dinv[grow];
            size_t o = (size_t)grow * GDH + col;
            z[o]   = val;
            agg[o] = val * di * di + bcol;               // self-loop + bias seed
        }
    }
}

// ---------------- edge scatter: agg[dst] += z[src] * dinv[src]*dinv[dst] ----
// One thread = one edge x 4 channels (b128 read, 4 L2-resident f32 atomics).

__global__ void k_scatter(const int* __restrict__ ei, const float* __restrict__ z,
                          const float* __restrict__ dinv, float* __restrict__ agg) {
    int idx = blockIdx.x * blockDim.x + threadIdx.x;     // E * 32
    if (idx < GE * (GDH / 4)) {
        int e = idx >> 5, q = idx & 31;
        int src = ei[e];
        int dst = ei[GE + e];
        float nrm = dinv[src] * dinv[dst];
        float4 zv = ((const float4*)(z + (size_t)src * GDH))[q];
        float* ap = agg + (size_t)dst * GDH + (q << 2);
        unsafeAtomicAdd(ap + 0, zv.x * nrm);             // global_atomic_add_f32
        unsafeAtomicAdd(ap + 1, zv.y * nrm);
        unsafeAtomicAdd(ap + 2, zv.z * nrm);
        unsafeAtomicAdd(ap + 3, zv.w * nrm);
    }
}

// ---------------- head MLP (single block, 128 threads) ----------------

__global__ void __launch_bounds__(128)
k_head(const float* __restrict__ agg, const int* __restrict__ nidx,
       const float* __restrict__ Wp, const float* __restrict__ bp,
       const float* __restrict__ Wa, const float* __restrict__ ba,
       const float* __restrict__ Wm, const float* __restrict__ bm,
       const float* __restrict__ Wg, const float* __restrict__ bg,
       const float* __restrict__ Wt, const float* __restrict__ bt,
       float* __restrict__ out) {
    __shared__ float sN[GDH];
    __shared__ float sH[GDH];
    int j = threadIdx.x;
    int ni = nidx[0];
    sN[j] = fmaxf(agg[(size_t)ni * GDH + j], 0.0f);      // relu(conv3 output row)
    __syncthreads();

    float acc = bp[j];
#pragma unroll 4
    for (int k = 0; k < GDH; ++k) acc += sN[k] * Wp[k * GDH + j];
    sH[j] = fmaxf(acc, 0.0f);
    __syncthreads();

    if (j < 30) {
        float o;
        if (j < 5) {
            o = ba[j];
            for (int k = 0; k < GDH; ++k) o += sH[k] * Wa[k * 5 + j];
        } else if (j < 7) {
            int jj = j - 5; o = bm[jj];
            for (int k = 0; k < GDH; ++k) o += sH[k] * Wm[k * 2 + jj];
        } else if (j < 10) {
            int jj = j - 7; o = bg[jj];
            for (int k = 0; k < GDH; ++k) o += sH[k] * Wg[k * 3 + jj];
        } else {
            int jj = j - 10; o = bt[jj];
            for (int k = 0; k < GDH; ++k) o += sH[k] * Wt[k * 20 + jj];
        }
        out[j] = o;
    }
}

// ---------------- launch ----------------

extern "C" void kernel_launch(void* const* d_in, const int* in_sizes, int n_in,
                              void* d_out, int out_size, void* d_ws, size_t ws_size,
                              hipStream_t stream) {
    const float* x   = (const float*)d_in[0];
    const int*   ei  = (const int*)d_in[1];
    const int*   nid = (const int*)d_in[2];
    const float* W1  = (const float*)d_in[3];
    const float* b1  = (const float*)d_in[4];
    const float* W2  = (const float*)d_in[5];
    const float* b2  = (const float*)d_in[6];
    const float* W3  = (const float*)d_in[7];
    const float* b3  = (const float*)d_in[8];
    const float* Wp  = (const float*)d_in[9];
    const float* bp  = (const float*)d_in[10];
    const float* Wa  = (const float*)d_in[11];
    const float* ba  = (const float*)d_in[12];
    const float* Wm  = (const float*)d_in[13];
    const float* bm  = (const float*)d_in[14];
    const float* Wg  = (const float*)d_in[15];
    const float* bg  = (const float*)d_in[16];
    const float* Wt  = (const float*)d_in[17];
    const float* bt  = (const float*)d_in[18];
    float* out = (float*)d_out;

    char* ws = (char*)d_ws;
    uint32_t* cnt  = (uint32_t*)(ws + 0);                        // 400 KB
    float*    dinv = (float*)   (ws + (1u << 19));               // 400 KB
    _Float16* wh   = (_Float16*)(ws + (1u << 20));               // 32 KB
    float*    z    = (float*)   (ws + (2ull  << 20));            // 51.2 MB
    float*    aggA = (float*)   (ws + (54ull << 20));            // 51.2 MB
    float*    aggB = (float*)   (ws + (106ull << 20));           // 51.2 MB

    const int T = 256;
    dim3 blk(T);

    // degree -> dinv
    hipMemsetAsync(cnt, 0, GN * sizeof(uint32_t), stream);
    k_count<<<(GE + T - 1) / T, blk, 0, stream>>>(ei, cnt);
    k_dinv<<<(GN + T - 1) / T, blk, 0, stream>>>(cnt, dinv);

    const int gGemm = (GN + 15) / 16;
    const int gE4   = (GE * (GDH / 4) + T - 1) / T;

    // ---- layer 1 (K = 64, input x, no relu on input) ----
    k_pack_w<<<(GDH * GDIN + T - 1) / T, blk, 0, stream>>>(W1, wh, GDIN);
    k_gemm<<<gGemm, blk, 0, stream>>>(x, GDIN, 0, wh, dinv, b1, z, aggA, GN);
    k_scatter<<<gE4, blk, 0, stream>>>(ei, z, dinv, aggA);

    // ---- layer 2 (K = 128, input relu(aggA)) ----
    k_pack_w<<<(GDH * GDH + T - 1) / T, blk, 0, stream>>>(W2, wh, GDH);
    k_gemm<<<gGemm, blk, 0, stream>>>(aggA, GDH, 1, wh, dinv, b2, z, aggB, GN);
    k_scatter<<<gE4, blk, 0, stream>>>(ei, z, dinv, aggB);

    // ---- layer 3 (K = 128, input relu(aggB)) ----
    k_pack_w<<<(GDH * GDH + T - 1) / T, blk, 0, stream>>>(W3, wh, GDH);
    k_gemm<<<gGemm, blk, 0, stream>>>(aggB, GDH, 1, wh, dinv, b3, z, aggA, GN);
    k_scatter<<<gE4, blk, 0, stream>>>(ei, z, dinv, aggA);

    // ---- head ----
    k_head<<<1, dim3(128), 0, stream>>>(aggA, nid, Wp, bp, Wa, ba, Wm, bm,
                                        Wg, bg, Wt, bt, out);
}
